// Euclidean_25649544691929
// MI455X (gfx1250) — matrix operand order repaired and checked by reading, not compiled, not run on hardware
//
#include <hip/hip_runtime.h>
#include <hip/hip_bf16.h>
#include <math.h>

// Problem constants (match reference)
#define BDIM 2048
#define IDIM 1024
#define ODIM 16384

// Tiling
#define BLK_M 128
#define BLK_N 128
#define BLK_K 32
#define BKP   40   // padded K-stride (elements) for B tile; 40*2B = 80B rows, 16B-aligned

typedef __attribute__((ext_vector_type(16))) __bf16        bf16x16;
typedef __attribute__((ext_vector_type(8)))  float         v8f;
typedef __attribute__((ext_vector_type(4)))  float         f32x4;
typedef __attribute__((ext_vector_type(4)))  unsigned int  u32x4;
typedef __attribute__((ext_vector_type(2)))  unsigned int  u32x2;
typedef unsigned short ushort_t;

union FragU {
    bf16x16 v;
    u32x4   q[2];
};

// ---- helpers -------------------------------------------------------------

static __device__ inline unsigned short bf16_rne(float f) {
    unsigned int u = __float_as_uint(f);
    u += 0x7FFFu + ((u >> 16) & 1u);           // round-to-nearest-even
    return (unsigned short)(u >> 16);
}
static __device__ inline float bf16_to_f32(unsigned short h) {
    return __uint_as_float(((unsigned int)h) << 16);
}
static __device__ inline void split_bf16(float f, unsigned short &hi, unsigned short &lo) {
    hi = bf16_rne(f);
    lo = bf16_rne(f - bf16_to_f32(hi));
}

static __device__ inline v8f wmma_bf16(const bf16x16 &a, const bf16x16 &b, v8f c) {
    // (neg_a, A, neg_b, B, c_mod, C, reuse_a, reuse_b)
    return __builtin_amdgcn_wmma_f32_16x16x32_bf16(false, a, false, b, (short)0, c, false, false);
}

// LDS byte offset of a generic pointer to a __shared__ object:
// flat LDS addresses carry the allocation-relative byte offset in the low 32 bits.
static __device__ inline unsigned lds_off(const void *p) {
    return (unsigned)(unsigned long long)p;
}

// 16 bytes per lane, global -> LDS, tracked with ASYNCcnt (gfx1250 async copy).
static __device__ inline void async_cp16(unsigned lds_byte_off, const void *gaddr) {
    asm volatile("global_load_async_to_lds_b128 %0, %1, off"
                 :: "v"(lds_byte_off), "v"((unsigned long long)gaddr)
                 : "memory");
}

// ---- preprocessing kernels ----------------------------------------------

// Split x (fp32) into bf16 hi/lo arrays, same [B][I] layout.
__global__ __launch_bounds__(256) void euclid_split_x(const float *__restrict__ x,
                                                      ushort_t *__restrict__ xhi,
                                                      ushort_t *__restrict__ xlo) {
    const size_t idx4 = (size_t)blockIdx.x * 256 + threadIdx.x;  // 2048 blocks: 2M/4 slots
    f32x4 f = *(const f32x4 *)(x + idx4 * 4);
    float vv[4] = {f.x, f.y, f.z, f.w};
    unsigned short h[4], l[4];
    for (int e = 0; e < 4; ++e) split_bf16(vv[e], h[e], l[e]);
    u32x2 ph, pl;
    ph.x = (unsigned int)h[0] | ((unsigned int)h[1] << 16);
    ph.y = (unsigned int)h[2] | ((unsigned int)h[3] << 16);
    pl.x = (unsigned int)l[0] | ((unsigned int)l[1] << 16);
    pl.y = (unsigned int)l[2] | ((unsigned int)l[3] << 16);
    *(u32x2 *)(xhi + idx4 * 4) = ph;
    *(u32x2 *)(xlo + idx4 * 4) = pl;
}

// Transpose w [I][O] fp32 -> wT [O][I] bf16 hi/lo (32x32 LDS tiles).
__global__ __launch_bounds__(256) void euclid_split_wT(const float *__restrict__ w,
                                                       ushort_t *__restrict__ whiT,
                                                       ushort_t *__restrict__ wloT) {
    __shared__ float t[32][33];
    const int to = blockIdx.x;   // O/32 = 512
    const int ti = blockIdx.y;   // I/32 = 32
    for (int j = 0; j < 4; ++j) {
        int e = threadIdx.x + j * 256;
        int r = e >> 5, c = e & 31;
        t[r][c] = w[(size_t)(ti * 32 + r) * ODIM + to * 32 + c];
    }
    __syncthreads();
    for (int j = 0; j < 4; ++j) {
        int e = threadIdx.x + j * 256;
        int r = e >> 5, c = e & 31;
        float v = t[c][r];                     // transposed read
        unsigned short h, l;
        split_bf16(v, h, l);
        size_t idx = (size_t)(to * 32 + r) * IDIM + ti * 32 + c;
        whiT[idx] = h;
        wloT[idx] = l;
    }
}

__global__ __launch_bounds__(256) void euclid_rownorm(const float *__restrict__ x,
                                                      float *__restrict__ x2) {
    __shared__ float red[256];
    const int row = blockIdx.x;
    const float *p = x + (size_t)row * IDIM;
    float s = 0.0f;
    for (int i = threadIdx.x; i < IDIM; i += 256) {
        float v = p[i];
        s += v * v;
    }
    red[threadIdx.x] = s;
    __syncthreads();
    for (int off = 128; off > 0; off >>= 1) {
        if (threadIdx.x < off) red[threadIdx.x] += red[threadIdx.x + off];
        __syncthreads();
    }
    if (threadIdx.x == 0) x2[row] = red[0];
}

__global__ __launch_bounds__(256) void euclid_colnorm(const float *__restrict__ w,
                                                      float *__restrict__ w2) {
    const int o = blockIdx.x * 256 + threadIdx.x;   // 64 blocks * 256 = 16384
    float s = 0.0f;
    for (int k = 0; k < IDIM; ++k) {
        float v = w[(size_t)k * ODIM + o];
        s += v * v;
    }
    w2[o] = s;
}

// ---- main GEMM + epilogue ------------------------------------------------

__global__ __launch_bounds__(256) void euclid_gemm(const ushort_t *__restrict__ xhi,
                                                   const ushort_t *__restrict__ xlo,
                                                   const ushort_t *__restrict__ whiT,
                                                   const ushort_t *__restrict__ wloT,
                                                   const float *__restrict__ x2,
                                                   const float *__restrict__ w2,
                                                   float *__restrict__ out) {
    // Double-buffered bf16 hi/lo tiles (72 KB total, well under 320 KB/WGP)
    __shared__ __align__(16) unsigned short sAhi[2][BLK_M * BLK_K];
    __shared__ __align__(16) unsigned short sAlo[2][BLK_M * BLK_K];
    __shared__ __align__(16) unsigned short sBhi[2][BLK_N * BKP];
    __shared__ __align__(16) unsigned short sBlo[2][BLK_N * BKP];

    const int tid  = threadIdx.x;
    const int lane = tid & 31;
    const int wave = tid >> 5;
    const int wm   = wave & 3;   // 4 waves along M, 32 rows each
    const int wn   = wave >> 2;  // 2 waves along N, 64 cols each
    const int nl   = lane & 15;
    const int half = lane >> 4;

    const int rowM = blockIdx.y * BLK_M;
    const int colN = blockIdx.x * BLK_N;

    v8f acc[2][4];
    for (int mt = 0; mt < 2; ++mt)
        for (int nt = 0; nt < 4; ++nt)
            for (int r = 0; r < 8; ++r)
                acc[mt][nt][r] = 0.0f;

    // Per-thread copy slots: 512 16B-chunks per 128x32 tile -> 2 per thread per array.
    // 8 async ops per thread per stage (Ahi, Alo, Bhi, Blo x 2 chunks).
    auto load_stage = [&](int st, int kb) {
        for (int j = 0; j < 2; ++j) {
            int cid = tid + j * 256;       // 0..511
            int r = cid >> 2;              // row 0..127
            int c = cid & 3;               // 16B chunk 0..3 (8 elements)
            size_t ga = (size_t)(rowM + r) * IDIM + kb + 8 * c;
            async_cp16(lds_off(&sAhi[st][r * BLK_K + 8 * c]), xhi + ga);
            async_cp16(lds_off(&sAlo[st][r * BLK_K + 8 * c]), xlo + ga);
            size_t gb = (size_t)(colN + r) * IDIM + kb + 8 * c;
            async_cp16(lds_off(&sBhi[st][r * BKP + 8 * c]), whiT + gb);
            async_cp16(lds_off(&sBlo[st][r * BKP + 8 * c]), wloT + gb);
        }
    };

    load_stage(0, 0);

    const int NK = IDIM / BLK_K;  // 32
    for (int ikb = 0; ikb < NK; ++ikb) {
        const int cur = ikb & 1;
        if (ikb + 1 < NK) {
            load_stage(cur ^ 1, (ikb + 1) * BLK_K);
            // current stage's 8 ops are the oldest outstanding -> wait until <= 8 remain
            asm volatile("s_wait_asynccnt 8" ::: "memory");
        } else {
            asm volatile("s_wait_asynccnt 0" ::: "memory");
        }
        __syncthreads();   // whole tile resident (all waves waited on their own ops)

        // ---- B fragments (shared across both m-tiles) ----
        FragU bh[4], bl[4];
        for (int nt = 0; nt < 4; ++nt) {
            int n = wn * 64 + nt * 16 + nl;
            const u32x4 *ph = (const u32x4 *)&sBhi[cur][n * BKP + 8 * half];
            const u32x4 *pl = (const u32x4 *)&sBlo[cur][n * BKP + 8 * half];
            bh[nt].q[0] = ph[0];
            bh[nt].q[1] = ph[2];
            bl[nt].q[0] = pl[0];
            bl[nt].q[1] = pl[2];
        }

        for (int mt = 0; mt < 2; ++mt) {
            int m = wm * 32 + mt * 16 + nl;
            const u32x4 *pah = (const u32x4 *)&sAhi[cur][m * BLK_K + 8 * half];
            const u32x4 *pal = (const u32x4 *)&sAlo[cur][m * BLK_K + 8 * half];
            FragU ah, al;
            ah.q[0] = pah[0];
            ah.q[1] = pah[2];   // +32 bytes (K += 16)
            al.q[0] = pal[0];
            al.q[1] = pal[2];
            for (int nt = 0; nt < 4; ++nt) {
                // x·w ≈ xhi·whi + xhi·wlo + xlo·whi  (bf16x3 split)
                acc[mt][nt] = wmma_bf16(ah.v, bh[nt].v, acc[mt][nt]);
                acc[mt][nt] = wmma_bf16(ah.v, bl[nt].v, acc[mt][nt]);
                acc[mt][nt] = wmma_bf16(al.v, bh[nt].v, acc[mt][nt]);
            }
        }
        __syncthreads();   // all waves done reading buf[cur] before next iter overwrites it
    }

    // ---- epilogue: out = sqrt(max(x2 + w2 - 2*acc, eps)) ----
    for (int mt = 0; mt < 2; ++mt) {
        for (int nt = 0; nt < 4; ++nt) {
            int gn = colN + wn * 64 + nt * 16 + nl;
            float w2v = w2[gn];
            for (int r = 0; r < 8; ++r) {
                int gm = rowM + wm * 32 + mt * 16 + r + half * 8;
                float d2 = x2[gm] + w2v - 2.0f * acc[mt][nt][r];
                out[(size_t)gm * ODIM + gn] = sqrtf(fmaxf(d2, 1e-12f));
            }
        }
    }
}

// ---- launch --------------------------------------------------------------

extern "C" void kernel_launch(void *const *d_in, const int *in_sizes, int n_in,
                              void *d_out, int out_size, void *d_ws, size_t ws_size,
                              hipStream_t stream) {
    (void)in_sizes; (void)n_in; (void)out_size; (void)ws_size;
    const float *x = (const float *)d_in[0];     // [2048, 1024]
    const float *w = (const float *)d_in[1];     // [1024, 16384]
    float *out = (float *)d_out;                 // [2048, 16384]

    // Workspace layout (~72.1 MB):
    char *ws = (char *)d_ws;
    float *x2 = (float *)ws;                              //  8 KB
    float *w2 = x2 + BDIM;                                // 64 KB
    ushort_t *xhi  = (ushort_t *)(ws + 73728);            //  4 MB
    ushort_t *xlo  = xhi + (size_t)BDIM * IDIM;           //  4 MB
    ushort_t *whiT = xlo + (size_t)BDIM * IDIM;           // 32 MB
    ushort_t *wloT = whiT + (size_t)ODIM * IDIM;          // 32 MB

    euclid_split_x<<<(BDIM * IDIM) / (256 * 4), 256, 0, stream>>>(x, xhi, xlo);
    euclid_split_wT<<<dim3(ODIM / 32, IDIM / 32), 256, 0, stream>>>(w, whiT, wloT);
    euclid_rownorm<<<BDIM, 256, 0, stream>>>(x, x2);
    euclid_colnorm<<<ODIM / 256, 256, 0, stream>>>(w, w2);
    euclid_gemm<<<dim3(ODIM / BLK_N, BDIM / BLK_M), 256, 0, stream>>>(
        xhi, xlo, whiT, wloT, x2, w2, out);
}
